// Head_38053410243058
// MI455X (gfx1250) — matrix operand order repaired and checked by reading, not compiled, not run on hardware
//
#include <hip/hip_runtime.h>

// Fused single-head attention: q/k/v projections + causal softmax + weighted sum.
// x:[B,8,384] f32, Wq/Wk/Wv:[384,64] f32, out:[B,8,64] f32.
// Each wave handles 2 batches packed as one M=16 WMMA tile. f16 inputs / f32 accum.

#define WSTRIDE 392            // LDS k-stride for Wt (padded: bank-conflict-free)
#define WAVES   16
#define TPB     (WAVES * 32)
#define BPB     (WAVES * 2)    // batches per block
#define WV_SCR  4608           // per-wave scratch in halves (q 1024 + k 1024 + vt 2048 + wei 512)

typedef _Float16 v16h __attribute__((ext_vector_type(16)));
typedef _Float16 v8h  __attribute__((ext_vector_type(8)));
typedef float    v8f  __attribute__((ext_vector_type(8)));

__device__ __forceinline__ v16h ld16_pair(const _Float16* p0, const _Float16* p1) {
    v8h lo = *(const v8h*)p0;
    v8h hi = *(const v8h*)p1;
    return __builtin_shufflevector(lo, hi, 0,1,2,3,4,5,6,7,8,9,10,11,12,13,14,15);
}

__device__ __forceinline__ v8f wmma_f16(v16h a, v16h b, v8f c) {
    return __builtin_amdgcn_wmma_f32_16x16x32_f16(false, a, false, b, (short)0, c,
                                                  false, false);
}

__device__ __forceinline__ v16h cvt_a(float4 f0, float4 f1, float4 f2, float4 f3) {
    v16h a;
    a[0]=(_Float16)f0.x;  a[1]=(_Float16)f0.y;  a[2]=(_Float16)f0.z;  a[3]=(_Float16)f0.w;
    a[4]=(_Float16)f1.x;  a[5]=(_Float16)f1.y;  a[6]=(_Float16)f1.z;  a[7]=(_Float16)f1.w;
    a[8]=(_Float16)f2.x;  a[9]=(_Float16)f2.y;  a[10]=(_Float16)f2.z; a[11]=(_Float16)f2.w;
    a[12]=(_Float16)f3.x; a[13]=(_Float16)f3.y; a[14]=(_Float16)f3.z; a[15]=(_Float16)f3.w;
    return a;
}

__global__ void __launch_bounds__(TPB)
head_attn_kernel(const float* __restrict__ x,  const float* __restrict__ Wq,
                 const float* __restrict__ Wk, const float* __restrict__ Wv,
                 float* __restrict__ out, int nbatch)
{
    extern __shared__ _Float16 smem[];
    _Float16* wt = smem;                               // [192][WSTRIDE] f16, k-contiguous
    _Float16* wavebase = smem + 192 * WSTRIDE;

    const int tid  = threadIdx.x;
    const int lane = tid & 31;
    const int w    = tid >> 5;

    // ---- Phase A: stage W^T (f16) into LDS: wt[(p*64+h)*WSTRIDE + k] = W_p[k][h]
    for (int p = 0; p < 3; ++p) {
        const float* Wsrc = (p == 0) ? Wq : (p == 1) ? Wk : Wv;
        for (int idx = tid; idx < 384 * 64; idx += TPB) {
            int k = idx >> 6, h = idx & 63;                // coalesced read of W[k][h]
            wt[(p * 64 + h) * WSTRIDE + k] = (_Float16)Wsrc[idx];
        }
    }
    __syncthreads();

    // ---- per-wave LDS scratch
    _Float16* q_lds = wavebase + (size_t)w * WV_SCR;   // [16][64]
    _Float16* k_lds = q_lds + 16 * 64;                 // [16][64]
    _Float16* vt    = k_lds + 16 * 64;                 // [64][32] (v transposed, s-padded)
    _Float16* wei_l = vt + 64 * 32;                    // [16][32] (K-padded)

    // zero the padded regions (cols 16..31) once
    for (int i = lane; i < 64 * 16; i += 32) vt[(i >> 4) * 32 + 16 + (i & 15)] = (_Float16)0.f;
    for (int i = lane; i < 16 * 16; i += 32) wei_l[(i >> 4) * 32 + 16 + (i & 15)] = (_Float16)0.f;

    const int b0 = blockIdx.x * BPB + 2 * w;
    if (b0 + 1 < nbatch) {
        const int hl      = lane & 15;
        const int half_hi = lane >> 4;          // 0: lanes 0-15, 1: lanes 16-31
        const int a_off   = half_hi ? 8 : 0;    // A-frag K sub-offset (16-bit A layout)
        const int b_off   = half_hi ? 8 : 0;    // B-frag K sub-offset (16-bit B layout)
        const int rowoff  = half_hi * 8;        // C-layout: M = j (+8 for upper lanes)
        const size_t tok0 = (size_t)b0 * 8;     // 16 contiguous token rows
        const float* xr   = x + (tok0 + hl) * 384;

        v8f zacc;
        #pragma unroll
        for (int i = 0; i < 8; ++i) zacc[i] = 0.0f;

        v8f acc[12];                            // tiles: q[0..3] k[4..7] v[8..11]
        #pragma unroll
        for (int t = 0; t < 12; ++t) acc[t] = zacc;

        // Hoisted B-tile base pointers: per-ks delta (<=704B) folds into DS offsets.
        const _Float16* bps[12];
        #pragma unroll
        for (int t = 0; t < 12; ++t)
            bps[t] = wt + (size_t)(t * 16 + hl) * WSTRIDE + b_off;

        // ---- Phase B: QKV projections. 12 K-steps of 32, 12 output tiles.
        // Software-pipelined: x loads for step ks+1 issue before the WMMA chain of ks;
        // all 12 B fragments are batch-loaded so DS latency overlaps the WMMA chain.
        const float* xp0 = xr + a_off;
        float4 f0 = *(const float4*)(xp0);
        float4 f1 = *(const float4*)(xp0 + 4);
        float4 f2 = *(const float4*)(xp0 + 16);
        float4 f3 = *(const float4*)(xp0 + 20);

        #pragma unroll
        for (int ks = 0; ks < 12; ++ks) {
            const int kb = ks * 32;
            v16h a = cvt_a(f0, f1, f2, f3);
            if (ks < 11) {                                // prefetch next K-step of x
                const float* xp = xr + kb + 32 + a_off;
                f0 = *(const float4*)(xp);
                f1 = *(const float4*)(xp + 4);
                f2 = *(const float4*)(xp + 16);
                f3 = *(const float4*)(xp + 20);
            }
            v16h bfr[12];
            #pragma unroll
            for (int t = 0; t < 12; ++t)
                bfr[t] = ld16_pair(bps[t] + kb, bps[t] + kb + 16);
            #pragma unroll
            for (int t = 0; t < 12; ++t)
                acc[t] = wmma_f16(a, bfr[t], acc[t]);
        }

        // ---- Phase C: spill q,k (row-major) and v (transposed) to wave LDS as f16
        #pragma unroll
        for (int nt = 0; nt < 4; ++nt) {
            v8h vpack;
            #pragma unroll
            for (int j = 0; j < 8; ++j) {
                q_lds[(j + rowoff) * 64 + nt * 16 + hl] = (_Float16)acc[nt][j];
                k_lds[(j + rowoff) * 64 + nt * 16 + hl] = (_Float16)acc[4 + nt][j];
                vpack[j] = (_Float16)acc[8 + nt][j];
            }
            *(v8h*)(vt + (nt * 16 + hl) * 32 + rowoff) = vpack;  // vt[h][s], 16B store
        }
        asm volatile("s_wait_dscnt 0x0" ::: "memory");

        // ---- Phase D: scores = q @ k^T  (16x16 tile; diagonal 8x8 blocks are valid)
        v8f wacc = zacc;
        #pragma unroll
        for (int ks = 0; ks < 2; ++ks) {
            const _Float16* qa = q_lds + hl * 64 + ks * 32 + a_off;
            v16h aq = ld16_pair(qa, qa + 16);
            const _Float16* kp = k_lds + hl * 64 + ks * 32 + b_off;
            v16h bk = ld16_pair(kp, kp + 16);
            wacc = wmma_f16(aq, bk, wacc);
        }

        // ---- Phase E: causal mask + softmax over 8 lanes (groups of 8, xor butterfly)
        const float scale = 0.051031036307982884f;   // 384^-0.5 (faithful to reference)
        const int sl = lane & 7;
        float wf[8];
        #pragma unroll
        for (int j = 0; j < 8; ++j) {
            float xv = wacc[j] * scale;
            xv = (sl <= j) ? xv : -1.0e30f;
            float m = xv;
            m = fmaxf(m, __shfl_xor(m, 1, 32));
            m = fmaxf(m, __shfl_xor(m, 2, 32));
            m = fmaxf(m, __shfl_xor(m, 4, 32));
            float p = __expf(xv - m);
            float s = p;
            s += __shfl_xor(s, 1, 32);
            s += __shfl_xor(s, 2, 32);
            s += __shfl_xor(s, 4, 32);
            wf[j] = p / s;
        }

        // ---- Phase F: write weights tile (zero cross-batch blocks; K 16..31 pre-zeroed)
        const bool inb = (lane < 8) || (lane >= 24);
        #pragma unroll
        for (int j = 0; j < 8; ++j)
            wei_l[(j + rowoff) * 32 + hl] = (_Float16)(inb ? wf[j] : 0.f);
        asm volatile("s_wait_dscnt 0x0" ::: "memory");

        // ---- Phase G: out = wei @ v, store f32 coalesced
        const _Float16* wa = wei_l + hl * 32 + a_off;
        v16h aw = ld16_pair(wa, wa + 16);
        #pragma unroll
        for (int nt = 0; nt < 4; ++nt) {
            const _Float16* bv = vt + (nt * 16 + hl) * 32 + b_off;
            v16h bb = ld16_pair(bv, bv + 16);
            v8f o = wmma_f16(aw, bb, zacc);
            #pragma unroll
            for (int j = 0; j < 8; ++j)
                out[(tok0 + j + rowoff) * 64 + nt * 16 + hl] = o[j];
        }
    }
}

extern "C" void kernel_launch(void* const* d_in, const int* in_sizes, int n_in,
                              void* d_out, int out_size, void* d_ws, size_t ws_size,
                              hipStream_t stream) {
    const float* x  = (const float*)d_in[0];
    const float* Wq = (const float*)d_in[1];
    const float* Wk = (const float*)d_in[2];
    const float* Wv = (const float*)d_in[3];
    float* out = (float*)d_out;

    int nbatch = in_sizes[0] / (8 * 384);
    int grid   = (nbatch + BPB - 1) / BPB;
    size_t shmem = (size_t)(192 * WSTRIDE + WAVES * WV_SCR) * 2;  // halves -> bytes (~291 KB)

    hipLaunchKernelGGL(head_attn_kernel, dim3(grid), dim3(TPB), shmem, stream,
                       x, Wq, Wk, Wv, out, nbatch);
}
